// MultiScaleRetention_2241972928527
// MI455X (gfx1250) — matrix operand order repaired and checked by reading, not compiled, not run on hardware
//
#include <hip/hip_runtime.h>

// ---------------- problem constants ----------------
#define B_   2
#define T_   2048
#define E_   1024
#define V_   2048
#define H_   8
#define DK_  128
#define DV_  256
#define M_   (B_ * T_)        // 4096 rows
#define EPS_ 1e-6f

// ---------------- WMMA types ----------------
typedef __attribute__((ext_vector_type(16))) __bf16 v16bf;
typedef __attribute__((ext_vector_type(8)))  float  v8f;

union FragU {
    v16bf v;
    uint4 u[2];
};

static __device__ __forceinline__ v8f wmma_bf16(const FragU& a, const FragU& b, v8f c) {
    return __builtin_amdgcn_wmma_f32_16x16x32_bf16(
        /*neg_a=*/false, a.v, /*neg_b=*/false, b.v,
        /*c_mod=*/(short)0, c, /*reuse_a=*/false, /*reuse_b=*/false);
}

// fp32 -> bf16 round-to-nearest-even
static __device__ __forceinline__ unsigned short f2bf(float f) {
    unsigned int u = __float_as_uint(f);
    u += 0x7FFFu + ((u >> 16) & 1u);
    return (unsigned short)(u >> 16);
}

// ---------------- Tensor Data Mover: 2D tile global -> LDS ----------------
// D# per CDNA5 ISA ch.8: group0 {count|lds_addr|global_addr|type=2},
// group1 {data_size=2B, tensor dims, tile dims, row stride}.
// clang-23 toolchain: 6-arg builtin (g0, g1, v4i, v4i, v8i, cpol).
typedef __attribute__((ext_vector_type(4))) unsigned int tdm_v4u;
typedef __attribute__((ext_vector_type(8))) int          tdm_v8i;
typedef __attribute__((ext_vector_type(4))) int          tdm_v4i;

static __device__ __forceinline__ unsigned lds_off_of(const void* p) {
    return (unsigned)(unsigned long long)p;   // low 32 bits of generic addr = LDS offset
}

// tile_w elements (2B each) per row, tile_h rows, row_stride in elements
static __device__ __forceinline__ void tdm_load_2d(unsigned lds_off, const void* gptr,
                                                   unsigned tile_w, unsigned tile_h,
                                                   unsigned row_stride) {
    unsigned long long ga = (unsigned long long)gptr;
    tdm_v4u g0;
    g0[0] = 1u;                                            // count=1, user descriptor
    g0[1] = lds_off;                                       // LDS byte address
    g0[2] = (unsigned)(ga & 0xFFFFFFFFu);                  // global_addr[31:0]
    g0[3] = (unsigned)((ga >> 32) & 0x1FFFFFFu) | 0x80000000u; // addr[56:32] | type=2
    tdm_v8i g1;
    g1[0] = (int)(1u << 16);                               // data_size=1 (2 bytes)
    g1[1] = (int)((tile_w & 0xFFFFu) << 16);               // tensor_dim0[15:0]
    g1[2] = (int)((tile_w >> 16) | ((tile_h & 0xFFFFu) << 16)); // td0 hi | tensor_dim1 lo
    g1[3] = (int)((tile_h >> 16) | ((tile_w & 0xFFFFu) << 16)); // td1 hi | tile_dim0
    g1[4] = (int)(tile_h & 0xFFFFu);                       // tile_dim1 (tile_dim2=0)
    g1[5] = (int)row_stride;                               // tensor_dim0_stride[31:0]
    g1[6] = 0;                                             // stride0 hi | stride1 lo
    g1[7] = 0;
    tdm_v4i z4 = {0, 0, 0, 0};
    tdm_v8i z8 = {0, 0, 0, 0, 0, 0, 0, 0};
    __builtin_amdgcn_tensor_load_to_lds(g0, g1, z4, z4, z8, 0);
}

// ---------------- elementwise fp32 -> bf16 (x4 vectorized) ----------------
__global__ __launch_bounds__(256) void cvt_bf16_4(const float* __restrict__ s,
                                                  unsigned short* __restrict__ d, int n4) {
    int i = blockIdx.x * 256 + threadIdx.x;
    if (i < n4) {
        float4 f = ((const float4*)s)[i];
        ushort4 o;
        o.x = f2bf(f.x); o.y = f2bf(f.y); o.z = f2bf(f.z); o.w = f2bf(f.w);
        ((ushort4*)d)[i] = o;
    }
}

// ---------------- generic GEMM: C[M,N] = A[M,K](bf16) * W[N,K](bf16)^T ----------------
// block tile 128x128, K-step 32; 8 waves, each owns 32 rows x 64 cols (8 WMMA/K-step).
// A/B tiles staged by TDM, double buffered, depth-1 prefetch.
__global__ __launch_bounds__(256) void gemm_bf16_nt(const unsigned short* __restrict__ A,
                                                    const unsigned short* __restrict__ W,
                                                    float* __restrict__ C,
                                                    int M, int N, int K) {
    __shared__ __align__(16) unsigned short sA[2][128 * 32];
    __shared__ __align__(16) unsigned short sB[2][128 * 32];

    const int tid  = threadIdx.x;
    const int lane = tid & 31;
    const int wave = tid >> 5;
    const int wm   = wave & 3;    // 32-row tile
    const int wn   = wave >> 2;   // 64-col group
    const int l15  = lane & 15;
    const int lh   = lane >> 4;
    const int bm   = blockIdx.y * 128;
    const int bn   = blockIdx.x * 128;

    v8f acc[2][4] = {};
    const int nk = K / 32;

    if (wave == 0) {
        tdm_load_2d(lds_off_of(sA[0]), A + (size_t)bm * K, 32, 128, (unsigned)K);
        tdm_load_2d(lds_off_of(sB[0]), W + (size_t)bn * K, 32, 128, (unsigned)K);
    }

    for (int s = 0; s < nk; s++) {
        const int cur = s & 1;
        if (wave == 0) {
            if (s + 1 < nk) {   // prefetch next tile pair into other buffer
                tdm_load_2d(lds_off_of(sA[cur ^ 1]), A + (size_t)bm * K + (s + 1) * 32,
                            32, 128, (unsigned)K);
                tdm_load_2d(lds_off_of(sB[cur ^ 1]), W + (size_t)bn * K + (s + 1) * 32,
                            32, 128, (unsigned)K);
                __builtin_amdgcn_s_wait_tensorcnt(2);   // current pair landed
            } else {
                __builtin_amdgcn_s_wait_tensorcnt(0);
            }
        }
        __syncthreads();

        const unsigned short* cA = sA[cur];
        const unsigned short* cB = sB[cur];
        FragU a0, a1;   // A layout: chunks at K {lh*8, 16+lh*8}
        a0.u[0] = *(const uint4*)(cA + (wm * 32 + l15) * 32 + lh * 8);
        a0.u[1] = *(const uint4*)(cA + (wm * 32 + l15) * 32 + 16 + lh * 8);
        a1.u[0] = *(const uint4*)(cA + (wm * 32 + 16 + l15) * 32 + lh * 8);
        a1.u[1] = *(const uint4*)(cA + (wm * 32 + 16 + l15) * 32 + 16 + lh * 8);
        for (int nt = 0; nt < 4; nt++) {
            int col = wn * 64 + nt * 16 + l15;
            FragU b;   // B layout: chunks at K {lh*8, 16+lh*8}
            b.u[0] = *(const uint4*)(cB + col * 32 + lh * 8);
            b.u[1] = *(const uint4*)(cB + col * 32 + 16 + lh * 8);
            acc[0][nt] = wmma_bf16(a0, b, acc[0][nt]);
            acc[1][nt] = wmma_bf16(a1, b, acc[1][nt]);
        }
        __syncthreads();   // reads of buf[cur] done before it is re-targeted
    }

    for (int mt = 0; mt < 2; mt++)
        for (int nt = 0; nt < 4; nt++) {
            int col = bn + wn * 64 + nt * 16 + l15;
            for (int j = 0; j < 8; j++) {
                int row = bm + wm * 32 + mt * 16 + j + 8 * lh;
                C[(size_t)row * N + col] = acc[mt][nt][j];
            }
        }
}

// ---------------- RoPE for q and k, fp32 [B,T,H,DK] -> bf16 [B,H,T,DK] ----------------
__global__ __launch_bounds__(256) void rope_qk(const float* __restrict__ q,
                                               const float* __restrict__ k,
                                               const float* __restrict__ sinp,
                                               const float* __restrict__ cosp,
                                               unsigned short* __restrict__ qr,
                                               unsigned short* __restrict__ kr) {
    int p = blockIdx.x * 256 + threadIdx.x;           // pair id
    if (p >= M_ * E_ / 2) return;
    int m = p / (E_ / 2);
    int e = (p % (E_ / 2)) * 2;
    int b = m / T_, t = m % T_;
    int hh = e / DK_, d = e % DK_;
    float s0 = sinp[t * DK_ + d], s1 = sinp[t * DK_ + d + 1];
    float c0 = cosp[t * DK_ + d], c1 = cosp[t * DK_ + d + 1];
    size_t src = (size_t)m * E_ + e;
    size_t dst = (((size_t)(b * H_ + hh)) * T_ + t) * DK_ + d;
    float x0 = q[src], x1 = q[src + 1];
    qr[dst]     = f2bf(x0 * c0 - x1 * s0);
    qr[dst + 1] = f2bf(x1 * c1 + x0 * s1);
    const float sc = 0.08838834764831845f;            // DK^-0.5
    x0 = k[src] * sc; x1 = k[src + 1] * sc;
    kr[dst]     = f2bf(x0 * c0 - x1 * s0);
    kr[dst + 1] = f2bf(x1 * c1 + x0 * s1);
}

// ---------------- V reorder: fp32 [B,T,H,DV] -> bf16 [B,H,DV,T] (LDS transpose) ------
__global__ __launch_bounds__(256) void reorder_v(const float* __restrict__ v,
                                                 unsigned short* __restrict__ vT) {
    __shared__ unsigned short tile[32][33];
    int bh = blockIdx.z;                  // b*H + h
    int tt = blockIdx.y * 32;
    int dd = blockIdx.x * 32;
    int b = bh / H_, hh = bh % H_;
    for (int i = threadIdx.x; i < 32 * 32; i += 256) {
        int lt = i / 32, ld = i % 32;
        float val = v[((size_t)(b * T_ + tt + lt)) * V_ + hh * DV_ + dd + ld];
        tile[ld][lt] = f2bf(val);
    }
    __syncthreads();
    for (int i = threadIdx.x; i < 32 * 32; i += 256) {
        int ld = i / 32, lt = i % 32;
        vT[(((size_t)bh * DV_) + dd + ld) * T_ + tt + lt] = tile[ld][lt];
    }
}

// ---------------- fused retention: S=qr.kr^T*mask, O=S.V/denom, RMS, SiLU gate -------
// grid: (T/64, B*H); 256 threads = 8 waves. Wave (wm 0..3, wn 0..1).
// sQ staged once by TDM; kr tiles double-buffered TDM with depth-1 prefetch.
__global__ __launch_bounds__(256) void retention_fused(
    const unsigned short* __restrict__ qr,   // [B,H,T,DK] bf16
    const unsigned short* __restrict__ kr,   // [B,H,T,DK] bf16
    const unsigned short* __restrict__ vT,   // [B,H,DV,T] bf16
    const float*          __restrict__ mask, // [H,T,T]
    const float*          __restrict__ gproj,// [M,V] fp32
    unsigned short*       __restrict__ gated)// [M,V] bf16
{
    __shared__ __align__(16) unsigned short sQ[64 * DK_];      // 16 KB
    __shared__ __align__(16) unsigned short sK[2][32 * DK_];   // 2 x 8 KB
    __shared__ __align__(16) unsigned short sP[64 * 32];       // 4 KB
    __shared__ float sRow[64];
    __shared__ float sSS[64];
    __shared__ float sInv[64];

    const int tid  = threadIdx.x;
    const int lane = tid & 31;
    const int wave = tid >> 5;
    const int wm   = wave & 3;
    const int wn   = wave >> 2;
    const int l15  = lane & 15;
    const int lh   = lane >> 4;
    const int bh   = blockIdx.y;
    const int b    = bh / H_, hh = bh % H_;
    const int q0   = blockIdx.x * 64;

    const size_t qbase = ((size_t)bh * T_ + q0) * DK_;
    const size_t kbase = (size_t)bh * T_ * DK_;
    const size_t vbase = (size_t)bh * DV_ * T_;
    const int nk = T_ / 32;

    if (wave == 0) {
        tdm_load_2d(lds_off_of(sQ), qr + qbase, DK_, 64, DK_);
        tdm_load_2d(lds_off_of(sK[0]), kr + kbase, DK_, 32, DK_);
    }
    if (tid < 64) { sRow[tid] = 0.f; sSS[tid] = 0.f; }

    v8f accO[8] = {};
    float srow[8] = {0.f, 0.f, 0.f, 0.f, 0.f, 0.f, 0.f, 0.f};

    for (int s = 0; s < nk; s++) {
        const int cur = s & 1;
        const int kt = s * 32;
        if (wave == 0) {
            if (s + 1 < nk) {
                tdm_load_2d(lds_off_of(sK[cur ^ 1]), kr + kbase + (size_t)(kt + 32) * DK_,
                            DK_, 32, DK_);
                __builtin_amdgcn_s_wait_tensorcnt(1);   // sQ + kr(s) landed
            } else {
                __builtin_amdgcn_s_wait_tensorcnt(0);
            }
        }
        __syncthreads();

        // S tile: this wave computes rows wm*16.., key-cols wn*16..
        const unsigned short* cK = sK[cur];
        v8f accS = {};
        for (int kc = 0; kc < DK_; kc += 32) {
            FragU a, bb;
            a.u[0]  = *(const uint4*)(sQ + (wm * 16 + l15) * DK_ + kc + lh * 8);
            a.u[1]  = *(const uint4*)(sQ + (wm * 16 + l15) * DK_ + kc + 16 + lh * 8);
            bb.u[0] = *(const uint4*)(cK + (wn * 16 + l15) * DK_ + kc + lh * 8);
            bb.u[1] = *(const uint4*)(cK + (wn * 16 + l15) * DK_ + kc + 16 + lh * 8);
            accS = wmma_bf16(a, bb, accS);
        }

        // decay mask, row-sum partials, write P (bf16) to LDS
        const int colg = kt + wn * 16 + l15;
        for (int j = 0; j < 8; j++) {
            int rowl = wm * 16 + j + 8 * lh;
            int rowg = q0 + rowl;
            float sv = accS[j] * mask[((size_t)hh * T_ + rowg) * T_ + colg];
            srow[j] += sv;
            sP[rowl * 32 + wn * 16 + l15] = f2bf(sv);
        }
        __syncthreads();

        // O += P(64x32) . Vtile(32x256); wave owns rows wm*16.., dv-cols wn*128..
        FragU pa;
        pa.u[0] = *(const uint4*)(sP + (wm * 16 + l15) * 32 + lh * 8);
        pa.u[1] = *(const uint4*)(sP + (wm * 16 + l15) * 32 + 16 + lh * 8);
        for (int nt = 0; nt < 8; nt++) {
            int dv = wn * 128 + nt * 16 + l15;
            FragU vb;   // B operand: t chunks at kt + {lh*8, 16+lh*8}
            vb.u[0] = *(const uint4*)(vT + vbase + (size_t)dv * T_ + kt + lh * 8);
            vb.u[1] = *(const uint4*)(vT + vbase + (size_t)dv * T_ + kt + 16 + lh * 8);
            accO[nt] = wmma_bf16(pa, vb, accO[nt]);
        }
    }

    // reduce row sums across the 16 lanes of each half, then across waves via LDS
    for (int j = 0; j < 8; j++) {
        float v = srow[j];
        for (int off = 1; off < 16; off <<= 1) v += __shfl_xor(v, off, 32);
        if (l15 == 0) atomicAdd(&sRow[wm * 16 + j + 8 * lh], v);
    }
    __syncthreads();
    if (tid < 64) {
        float d = fabsf(sRow[tid]);
        sInv[tid] = 1.f / (d < 1.f ? 1.f : d);
    }
    __syncthreads();

    // apply 1/denom, accumulate sum-of-squares for RMS over DV
    float ss[8] = {0.f, 0.f, 0.f, 0.f, 0.f, 0.f, 0.f, 0.f};
    for (int j = 0; j < 8; j++) {
        float inv = sInv[wm * 16 + j + 8 * lh];
        for (int nt = 0; nt < 8; nt++) {
            float o = accO[nt][j] * inv;
            accO[nt][j] = o;
            ss[j] += o * o;
        }
    }
    for (int j = 0; j < 8; j++) {
        float v = ss[j];
        for (int off = 1; off < 16; off <<= 1) v += __shfl_xor(v, off, 32);
        if (l15 == 0) atomicAdd(&sSS[wm * 16 + j + 8 * lh], v);
    }
    __syncthreads();
    if (tid < 64) sInv[tid] = rsqrtf(sSS[tid] / (float)DV_ + EPS_);
    __syncthreads();

    // RMS-norm + SiLU(g) gate, store bf16 [M,V]
    for (int nt = 0; nt < 8; nt++) {
        int dv = wn * 128 + nt * 16 + l15;
        int colV = hh * DV_ + dv;
        for (int j = 0; j < 8; j++) {
            int rowl = wm * 16 + j + 8 * lh;
            int t = q0 + rowl;
            size_t gi = ((size_t)(b * T_ + t)) * V_ + colV;
            float gv = gproj[gi];
            float silu = gv / (1.f + __expf(-gv));
            gated[gi] = f2bf(accO[nt][j] * sInv[rowl] * silu);
        }
    }
}

// ---------------- host-side orchestration ----------------
extern "C" void kernel_launch(void* const* d_in, const int* in_sizes, int n_in,
                              void* d_out, int out_size, void* d_ws, size_t ws_size,
                              hipStream_t stream) {
    (void)in_sizes; (void)n_in; (void)out_size; (void)ws_size;
    const float* hs    = (const float*)d_in[0];
    const float* w_q   = (const float*)d_in[1];
    const float* w_k   = (const float*)d_in[2];
    const float* w_v   = (const float*)d_in[3];
    const float* w_g   = (const float*)d_in[4];
    const float* w_out = (const float*)d_in[5];
    const float* sinp  = (const float*)d_in[6];
    const float* cosp  = (const float*)d_in[7];
    const float* dmask = (const float*)d_in[8];
    float* out = (float*)d_out;

    char* ws = (char*)d_ws;
    size_t off = 0;
    auto alloc = [&](size_t bytes) -> char* {
        char* p = ws + off;
        off += (bytes + 255) & ~(size_t)255;
        return p;
    };
    unsigned short* hs_b   = (unsigned short*)alloc((size_t)M_ * E_ * 2);
    unsigned short* wq_b   = (unsigned short*)alloc((size_t)E_ * E_ * 2);
    unsigned short* wk_b   = (unsigned short*)alloc((size_t)E_ * E_ * 2);
    unsigned short* wv_b   = (unsigned short*)alloc((size_t)V_ * E_ * 2);
    unsigned short* wg_b   = (unsigned short*)alloc((size_t)V_ * E_ * 2);
    unsigned short* wo_b   = (unsigned short*)alloc((size_t)E_ * V_ * 2);
    float*          q_f    = (float*)alloc((size_t)M_ * E_ * 4);
    float*          k_f    = (float*)alloc((size_t)M_ * E_ * 4);
    float*          v_f    = (float*)alloc((size_t)M_ * V_ * 4);
    float*          g_f    = (float*)alloc((size_t)M_ * V_ * 4);
    unsigned short* qr_b   = (unsigned short*)alloc((size_t)M_ * E_ * 2);
    unsigned short* kr_b   = (unsigned short*)alloc((size_t)M_ * E_ * 2);
    unsigned short* vT_b   = (unsigned short*)alloc((size_t)M_ * V_ * 2);
    unsigned short* gated  = (unsigned short*)alloc((size_t)M_ * V_ * 2);

    auto cvt = [&](const float* s, unsigned short* d, size_t n) {
        int n4 = (int)(n / 4);
        cvt_bf16_4<<<(n4 + 255) / 256, 256, 0, stream>>>(s, d, n4);
    };
    cvt(hs,    hs_b, (size_t)M_ * E_);
    cvt(w_q,   wq_b, (size_t)E_ * E_);
    cvt(w_k,   wk_b, (size_t)E_ * E_);
    cvt(w_v,   wv_b, (size_t)V_ * E_);
    cvt(w_g,   wg_b, (size_t)V_ * E_);
    cvt(w_out, wo_b, (size_t)E_ * V_);

    // projections: C[M,N] = hs[M,K] . W[N,K]^T
    gemm_bf16_nt<<<dim3(E_ / 128, M_ / 128), 256, 0, stream>>>(hs_b, wq_b, q_f, M_, E_, E_);
    gemm_bf16_nt<<<dim3(E_ / 128, M_ / 128), 256, 0, stream>>>(hs_b, wk_b, k_f, M_, E_, E_);
    gemm_bf16_nt<<<dim3(V_ / 128, M_ / 128), 256, 0, stream>>>(hs_b, wv_b, v_f, M_, V_, E_);
    gemm_bf16_nt<<<dim3(V_ / 128, M_ / 128), 256, 0, stream>>>(hs_b, wg_b, g_f, M_, V_, E_);

    // RoPE + head-major repack
    rope_qk<<<(M_ * E_ / 2 + 255) / 256, 256, 0, stream>>>(q_f, k_f, sinp, cosp, qr_b, kr_b);

    // V transpose to [B,H,DV,T]
    reorder_v<<<dim3(DV_ / 32, T_ / 32, B_ * H_), 256, 0, stream>>>(v_f, vT_b);

    // fused retention + RMS-norm + SiLU gating
    retention_fused<<<dim3(T_ / 64, B_ * H_), 256, 0, stream>>>(qr_b, kr_b, vT_b, dmask, g_f, gated);

    // output projection: out[M,E] = gated[M,V] . w_out[E,V]^T
    gemm_bf16_nt<<<dim3(E_ / 128, M_ / 128), 256, 0, stream>>>(gated, wo_b, out, M_, E_, V_);
}
